// GWKLayer_exp_39393440039434
// MI455X (gfx1250) — matrix operand order repaired and checked by reading, not compiled, not run on hardware
//
#include <hip/hip_runtime.h>
#include <hip/hip_bf16.h>

#define NT  8192   // N tokens
#define DIN 512    // input dim
#define DD  256    // head dim

typedef __attribute__((ext_vector_type(16))) __bf16 v16bf;
typedef __attribute__((ext_vector_type(8)))  float  v8f;
typedef __attribute__((ext_vector_type(8)))  unsigned short u16x8;
typedef __attribute__((ext_vector_type(4)))  float  f32x4;

union BF16x16 { v16bf v; unsigned short u[16]; u16x8 h[2]; };
union F32x8   { v8f v; float f[8]; };
union U16x8   { u16x8 v; unsigned short u[8]; };

__device__ __forceinline__ unsigned short f2bf(float f) {     // RNE (cold paths)
  unsigned int x = __float_as_uint(f);
  x += 0x7FFFu + ((x >> 16) & 1u);
  return (unsigned short)(x >> 16);
}
__device__ __forceinline__ unsigned short f2bf_rz(float f) {  // truncate (hot path)
  return (unsigned short)(__float_as_uint(f) >> 16);
}

// ---------------------------------------------------------------------------
// Prep 1: feat (fp32) -> featB (bf16 row-major). One streaming pass.
// ---------------------------------------------------------------------------
__global__ __launch_bounds__(256) void cvt_feat_kernel(
    const float* __restrict__ feat, unsigned short* __restrict__ featB)
{
  const size_t i = ((size_t)blockIdx.x * 256 + threadIdx.x) * 8;
  f32x4 a = *(const f32x4*)(feat + i);
  f32x4 b = *(const f32x4*)(feat + i + 4);
  U16x8 o;
  #pragma unroll
  for (int j = 0; j < 4; ++j) { o.u[j] = f2bf(a[j]); o.u[4 + j] = f2bf(b[j]); }
  *(u16x8*)(featB + i) = o.v;
}

// ---------------------------------------------------------------------------
// Prep 2: Wq/Wk/Wv [512][256] fp32 -> Wt [3][256][512] bf16 (transposed).
// The attention scale 1/sqrt(256) is folded into Wq here.
// ---------------------------------------------------------------------------
__global__ __launch_bounds__(256) void cvt_w_kernel(
    const float* __restrict__ Wq, const float* __restrict__ Wk,
    const float* __restrict__ Wv, unsigned short* __restrict__ Wt)
{
  const int idx = blockIdx.x * 256 + threadIdx.x;  // 3 * 256 * 64 threads
  const int kc = idx & 63;
  const int n  = (idx >> 6) & 255;
  const int which = idx >> 14;
  const float* W = (which == 0) ? Wq : (which == 1) ? Wk : Wv;
  const float s = (which == 0) ? 0.0625f : 1.0f;
  U16x8 o;
  #pragma unroll
  for (int j = 0; j < 8; ++j)
    o.u[j] = f2bf(W[(size_t)(kc * 8 + j) * DD + n] * s);
  *(u16x8*)(Wt + (size_t)which * DD * DIN + (size_t)n * DIN + kc * 8) = o.v;
}

// ---------------------------------------------------------------------------
// Kernel 1: Q/K (row-major bf16), Vt (transposed bf16) projections.
// One 16x32 output slab per wave; hot loop = 6 x b128 load + 2 x WMMA.
// ---------------------------------------------------------------------------
__global__ __launch_bounds__(128) void qkv_kernel(
    const unsigned short* __restrict__ featB, const unsigned short* __restrict__ Wt,
    unsigned short* __restrict__ Qb, unsigned short* __restrict__ Kb,
    unsigned short* __restrict__ Vt)
{
  const int lane = threadIdx.x & 31;
  const int wave = threadIdx.x >> 5;
  const int rowTile = blockIdx.x * 4 + wave;
  const int which   = blockIdx.z;
  const int m16 = lane & 15;
  const int g   = lane >> 4;
  const int rowbase = rowTile * 16;
  const int colbase = blockIdx.y * 32;

  v8f acc0 = {}, acc1 = {};
  const unsigned short* arow  = featB + (size_t)(rowbase + m16) * DIN;
  const unsigned short* brow0 = Wt + (size_t)which * DD * DIN + (size_t)(colbase + m16) * DIN;
  const unsigned short* brow1 = brow0 + (size_t)16 * DIN;

  #pragma unroll
  for (int kc = 0; kc < DIN / 32; ++kc) {
    const int k0 = kc * 32;
    BF16x16 a;
    a.h[0] = *(const u16x8*)(arow + k0 + 8 * g);
    a.h[1] = *(const u16x8*)(arow + k0 + 16 + 8 * g);
    BF16x16 b0, b1;
    b0.h[0] = *(const u16x8*)(brow0 + k0 + 8 * g);
    b0.h[1] = *(const u16x8*)(brow0 + k0 + 16 + 8 * g);
    b1.h[0] = *(const u16x8*)(brow1 + k0 + 8 * g);
    b1.h[1] = *(const u16x8*)(brow1 + k0 + 16 + 8 * g);
    acc0 = __builtin_amdgcn_wmma_f32_16x16x32_bf16(false, a.v, false, b0.v,
                                                   (short)0, acc0, false, false);
    acc1 = __builtin_amdgcn_wmma_f32_16x16x32_bf16(false, a.v, false, b1.v,
                                                   (short)0, acc1, false, false);
  }

  F32x8 c0, c1; c0.v = acc0; c1.v = acc1;
  if (which == 2) {
    #pragma unroll
    for (int r = 0; r < 8; ++r) {
      const int m = rowbase + 8 * g + r;
      Vt[(size_t)(colbase + m16) * NT + m]      = f2bf(c0.f[r]);
      Vt[(size_t)(colbase + 16 + m16) * NT + m] = f2bf(c1.f[r]);
    }
  } else {
    unsigned short* O = (which == 0) ? Qb : Kb;
    #pragma unroll
    for (int r = 0; r < 8; ++r) {
      const int m = rowbase + 8 * g + r;
      O[(size_t)m * DD + colbase + m16]      = f2bf(c0.f[r]);
      O[(size_t)m * DD + colbase + 16 + m16] = f2bf(c1.f[r]);
    }
  }
}

// ---------------------------------------------------------------------------
// Kernel 2: fused attention with counting_attn weighting + ELU.
// Max-subtraction cancels exactly for this operator (logits O(1), denominator
// O(10^3) >> 1e-9), so no online-max machinery.
// 512 blocks x 64 threads (2 waves). The two waves COOPERATE on S:
//   wave w computes S for keys j+16w..j+16w+15 (8 WMMAs), applies exp*Cnt to
//   its 16 P columns and writes them into a shared double-buffered LDS tile.
//   One __syncthreads() per step (double buffer => single barrier race-free),
//   then each wave reads the full 16x32 P in A layout and does its own
//   row-sum WMMA (P @ ones) + 8 P@V WMMAs for its 128 output columns.
// Per wave per 32-key step: 17 WMMAs, 8 exp, 8 pack, 8 Cnt loads -> the
// 268 MB counting_attn stream is read exactly once (the roofline floor).
// ---------------------------------------------------------------------------
__global__ __launch_bounds__(64) void attn_kernel(
    const unsigned short* __restrict__ Qb, const unsigned short* __restrict__ Kb,
    const unsigned short* __restrict__ Vt, const float* __restrict__ Cnt,
    float* __restrict__ Out)
{
  __shared__ unsigned short pstage[2][16][32];   // [buffer][row][key col]

  const int lane = threadIdx.x & 31;
  const int wave = threadIdx.x >> 5;         // 0/1: key half + output col half
  const int rowbase = blockIdx.x * 16;
  const int colbase = wave * 128;
  const int nl = lane & 15;
  const int g  = lane >> 4;

  // Q A-operands (16 rows x 256 k) hoisted into registers.
  v16bf qa[8];
  {
    const unsigned short* qrow = Qb + (size_t)(rowbase + nl) * DD;
    #pragma unroll
    for (int kc = 0; kc < 8; ++kc) {
      BF16x16 t;
      t.h[0] = *(const u16x8*)(qrow + kc * 32 + 8 * g);
      t.h[1] = *(const u16x8*)(qrow + kc * 32 + 16 + 8 * g);
      qa[kc] = t.v;
    }
  }

  BF16x16 ones;                               // B operand of all 1.0bf16
  #pragma unroll
  for (int j = 0; j < 16; ++j) ones.u[j] = 0x3F80;

  v8f o[8];
  v8f lacc = {};                              // row sums, replicated across cols
  #pragma unroll
  for (int t = 0; t < 8; ++t) o[t] = (v8f){};

  for (int j = 0; j < NT; j += 32) {
    const int buf = (j >> 5) & 1;

    // ---- this wave's S half: keys j+16w .. j+16w+15 (one 16x16 tile) -------
    v8f s = {};
    const unsigned short* krow = Kb + (size_t)(j + 16 * wave + nl) * DD;
    #pragma unroll
    for (int kc = 0; kc < 8; ++kc) {
      BF16x16 b;
      const u16x8* p = (const u16x8*)(krow + kc * 32 + 8 * g);
      b.h[0] = p[0]; b.h[1] = p[2];
      s = __builtin_amdgcn_wmma_f32_16x16x32_bf16(false, qa[kc], false, b.v,
                                                  (short)0, s, false, false);
    }
    F32x8 S; S.v = s;

    // ---- P = exp(S) * counting_attn for our 16 key columns -> shared LDS ---
    #pragma unroll
    for (int r = 0; r < 8; ++r) {
      const size_t row = (size_t)(rowbase + 8 * g + r);
      const float c = Cnt[row * NT + j + 16 * wave + nl];
      if (j + 256 < NT) __builtin_prefetch(&Cnt[row * NT + j + 256 + 16 * wave + nl], 0, 1);
      pstage[buf][8 * g + r][16 * wave + nl] = f2bf_rz(__expf(S.f[r]) * c);
    }
    __syncthreads();

    // ---- full 16x32 P in A layout ------------------------------------------
    BF16x16 pa;
    #pragma unroll
    for (int jj = 0; jj < 8; ++jj) pa.u[jj]     = pstage[buf][nl][8 * g + jj];
    #pragma unroll
    for (int jj = 0; jj < 8; ++jj) pa.u[8 + jj] = pstage[buf][nl][16 + 8 * g + jj];

    // ---- running row sums: lacc += P @ ones --------------------------------
    lacc = __builtin_amdgcn_wmma_f32_16x16x32_bf16(false, pa.v, false, ones.v,
                                                   (short)0, lacc, false, false);

    // ---- O += P @ V (this wave's 128 output columns) -----------------------
    #pragma unroll
    for (int t = 0; t < 8; ++t) {
      BF16x16 vb;
      const unsigned short* vcol =
          Vt + (size_t)(colbase + t * 16 + nl) * NT + j + 8 * g;
      vb.h[0] = *(const u16x8*)(vcol);
      vb.h[1] = *(const u16x8*)(vcol + 16);
      o[t] = __builtin_amdgcn_wmma_f32_16x16x32_bf16(false, pa.v, false, vb.v,
                                                     (short)0, o[t], false, false);
    }
  }

  // ---- finalize: divide by (l + 1e-9), ELU, store fp32 --------------------
  F32x8 L; L.v = lacc;
  #pragma unroll
  for (int t = 0; t < 8; ++t) {
    F32x8 ot; ot.v = o[t];
    #pragma unroll
    for (int r = 0; r < 8; ++r) {
      float v = ot.f[r] / (L.f[r] + 1e-9f);
      v = (v > 0.0f) ? v : (__expf(v) - 1.0f);
      const int m = rowbase + 8 * g + r;
      Out[(size_t)m * DD + colbase + t * 16 + nl] = v;
    }
  }
}

// ---------------------------------------------------------------------------
extern "C" void kernel_launch(void* const* d_in, const int* in_sizes, int n_in,
                              void* d_out, int out_size, void* d_ws, size_t ws_size,
                              hipStream_t stream) {
  (void)in_sizes; (void)n_in; (void)out_size; (void)ws_size;
  const float* feat = (const float*)d_in[0];
  const float* cnt  = (const float*)d_in[1];
  const float* Wq   = (const float*)d_in[2];
  const float* Wk   = (const float*)d_in[3];
  const float* Wv   = (const float*)d_in[4];
  float* out = (float*)d_out;

  unsigned short* Qb    = (unsigned short*)d_ws;            // 8192x256 bf16
  unsigned short* Kb    = Qb + (size_t)NT * DD;             // 8192x256 bf16
  unsigned short* Vt    = Kb + (size_t)NT * DD;             // 256x8192 bf16
  unsigned short* featB = Vt + (size_t)NT * DD;             // 8192x512 bf16
  unsigned short* Wt    = featB + (size_t)NT * DIN;         // 3x256x512 bf16

  cvt_feat_kernel<<<dim3((NT * DIN) / (256 * 8)), 256, 0, stream>>>(feat, featB);
  cvt_w_kernel<<<dim3((3 * DD * (DIN / 8)) / 256), 256, 0, stream>>>(Wq, Wk, Wv, Wt);
  qkv_kernel<<<dim3(128, 8, 3), 128, 0, stream>>>(featB, Wt, Qb, Kb, Vt);
  attn_kernel<<<dim3(512), 64, 0, stream>>>(Qb, Kb, Vt, cnt, out);
}
